// HybridTrojanNetPyGMultiTask_82798379532723
// MI455X (gfx1250) — matrix operand order repaired and checked by reading, not compiled, not run on hardware
//
#include <hip/hip_runtime.h>
#include <cstddef>
#include <cstdint>

#define NN 16384
#define BB 16
#define SS 1024
#define EE 262144
#define CC 256
#define HH 4
#define HDIM 64
#define LRELU_S 0.2f

#define GF_RELU 1
#define GF_ACCUM 2

typedef unsigned short u16;
typedef __attribute__((ext_vector_type(16))) __bf16 v16bf;
typedef __attribute__((ext_vector_type(8)))  float v8f;

union BF16x16 { u16 us[16]; unsigned u4[8]; v16bf v; };

__device__ __forceinline__ u16 bf16b(float f) {
  unsigned u = __builtin_bit_cast(unsigned, f);
  return (u16)((u + 0x7FFFu + ((u >> 16) & 1u)) >> 16);
}
__device__ __forceinline__ void cv4(const float4 f, u16* o) {
  o[0] = bf16b(f.x); o[1] = bf16b(f.y); o[2] = bf16b(f.z); o[3] = bf16b(f.w);
}
__device__ __forceinline__ void atomAddF(float* p, float v) {
  __hip_atomic_fetch_add(p, v, __ATOMIC_RELAXED, __HIP_MEMORY_SCOPE_AGENT);
}
// order-preserving float<->uint encoding for atomicMax-based segment max
__device__ __forceinline__ unsigned fenc(float f) {
  unsigned u = __builtin_bit_cast(unsigned, f);
  return u ^ ((u & 0x80000000u) ? 0xFFFFFFFFu : 0x80000000u);
}
__device__ __forceinline__ float fdec(unsigned k) {
  unsigned u = k ^ ((k & 0x80000000u) ? 0x80000000u : 0xFFFFFFFFu);
  return __builtin_bit_cast(float, u);
}

// A-operand fragment: 8 bf16 at pa[0..7], 8 at pa[16..23] (ISA A layout)
__device__ __forceinline__ v16bf ldfragA(const u16* pa) {
  BF16x16 f;
  *(uint4*)&f.u4[0] = *(const uint4*)(pa);
  *(uint4*)&f.u4[4] = *(const uint4*)(pa + 16);
  return f.v;
}
// B-operand fragment: one N column, K rows pb[0..7]*ld and pb[16..23]*ld (ISA layout)
__device__ __forceinline__ v16bf ldfragB(const u16* pb, size_t ld) {
  BF16x16 f;
#pragma unroll
  for (int s = 0; s < 8; ++s) { f.us[s] = pb[(size_t)s * ld]; f.us[8 + s] = pb[(size_t)(16 + s) * ld]; }
  return f.v;
}

// ---------------------------------------------------------------------------
// Streaming f32 -> bf16 (RNE) conversion; 8 elements/thread.
// ---------------------------------------------------------------------------
__global__ void k_cvt(const float* __restrict__ in, u16* __restrict__ out, long long n) {
  long long i = ((long long)blockIdx.x * 256 + threadIdx.x) * 8;
  if (i >= n) return;
  union { u16 us[8]; uint4 u; } t;
  cv4(*(const float4*)(in + i), t.us);
  cv4(*(const float4*)(in + i + 4), t.us + 4);
  *(uint4*)(out + i) = t.u;
}

// ---------------------------------------------------------------------------
// bf16-WMMA GEMM (software-pipelined): C = op(A16[MxK] @ W16[KxNc] (+bias)(+C))
// block = 128 threads = 4 waves; wave computes a 16x64 output strip.
// ---------------------------------------------------------------------------
__global__ __launch_bounds__(128) void k_gemm(
    const u16* __restrict__ A16, int lda,
    const u16* __restrict__ W16, int ldw,
    const float* __restrict__ bias,
    float* C, int ldc, int M, int K, int Nc, int flags) {
  int lane = threadIdx.x & 31;
  int wid  = threadIdx.x >> 5;
  int m0 = (blockIdx.x * 4 + wid) * 16;
  int n0base = blockIdx.y * 64;
  if (m0 >= M) return;
  int arow = m0 + (lane & 15);
  int kh = lane >> 4;
  v8f z = {};
  v8f acc[4];
  acc[0] = z; acc[1] = z; acc[2] = z; acc[3] = z;
  const u16* pa = A16 + (size_t)arow * lda + 8 * kh;
  const u16* pb = W16 + (size_t)(8 * kh) * ldw + n0base + (lane & 15);   // B: lane = n column
  const size_t bstep = (size_t)ldw * 32;
  // stage 0 preload
  v16bf av = ldfragA(pa);
  v16bf b0 = ldfragB(pb, ldw);
  v16bf b1 = ldfragB(pb + 16, ldw);
  v16bf b2 = ldfragB(pb + 32, ldw);
  v16bf b3 = ldfragB(pb + 48, ldw);
  for (int k0 = 32; k0 < K; k0 += 32) {
    pa += 32; pb += bstep;
    // prefetch next k-step while current WMMAs issue
    v16bf avn = ldfragA(pa);
    v16bf c0 = ldfragB(pb, ldw);
    v16bf c1 = ldfragB(pb + 16, ldw);
    v16bf c2 = ldfragB(pb + 32, ldw);
    v16bf c3 = ldfragB(pb + 48, ldw);
    acc[0] = __builtin_amdgcn_wmma_f32_16x16x32_bf16(false, av, false, b0, (short)0, acc[0], false, false);
    acc[1] = __builtin_amdgcn_wmma_f32_16x16x32_bf16(false, av, false, b1, (short)0, acc[1], false, false);
    acc[2] = __builtin_amdgcn_wmma_f32_16x16x32_bf16(false, av, false, b2, (short)0, acc[2], false, false);
    acc[3] = __builtin_amdgcn_wmma_f32_16x16x32_bf16(false, av, false, b3, (short)0, acc[3], false, false);
    av = avn; b0 = c0; b1 = c1; b2 = c2; b3 = c3;
  }
  acc[0] = __builtin_amdgcn_wmma_f32_16x16x32_bf16(false, av, false, b0, (short)0, acc[0], false, false);
  acc[1] = __builtin_amdgcn_wmma_f32_16x16x32_bf16(false, av, false, b1, (short)0, acc[1], false, false);
  acc[2] = __builtin_amdgcn_wmma_f32_16x16x32_bf16(false, av, false, b2, (short)0, acc[2], false, false);
  acc[3] = __builtin_amdgcn_wmma_f32_16x16x32_bf16(false, av, false, b3, (short)0, acc[3], false, false);
#pragma unroll
  for (int t = 0; t < 4; ++t) {
    int n = n0base + t * 16 + (lane & 15);
    float bvv = bias ? bias[n] : 0.0f;
#pragma unroll
    for (int r = 0; r < 8; ++r) {
      int m = m0 + 8 * kh + r;             // D layout: m = r + 8*(lane>>4)
      float v = acc[t][r] + bvv;
      float* cp = C + (size_t)m * ldc + n;
      if (flags & GF_ACCUM) v += *cp;
      if (flags & GF_RELU)  v = fmaxf(v, 0.0f);
      *cp = v;
    }
  }
}

// ---------------------------------------------------------------------------
// K transpose: qkv16 k-part -> kT[(b,h,d),key] so flash B-frags are vector loads
// ---------------------------------------------------------------------------
__global__ void k_ktrans(const u16* __restrict__ qkv16, u16* __restrict__ kT) {
  int gid = blockIdx.x * 256 + threadIdx.x;     // total BB*HH*64*128
  int kc = gid & 127;           // key chunk of 8
  int d  = (gid >> 7) & 63;
  int bh = gid >> 13;
  int b = bh >> 2, h = bh & 3;
  const u16* src = qkv16 + (size_t)(b * SS + kc * 8) * 768 + 256 + h * HDIM + d;
  union { u16 us[8]; uint4 u; } t;
#pragma unroll
  for (int j = 0; j < 8; ++j) t.us[j] = src[(size_t)j * 768];
  *(uint4*)(kT + ((size_t)bh * HDIM + d) * SS + kc * 8) = t.u;
}

// ---------------------------------------------------------------------------
// Flash attention over bf16 qkv: one wave per 16 queries of one (graph, head).
// qkv16: [N,768] bf16 (q|k|v); kT: [(b,h,d),key] bf16; o: [N,256] f32
// ---------------------------------------------------------------------------
__global__ __launch_bounds__(128) void k_flash(const u16* __restrict__ qkv16,
                                               const u16* __restrict__ kT,
                                               float* __restrict__ o) {
  __shared__ float s_sc[4][16][32];
  __shared__ u16 s_p[4][16][32];
  __shared__ float s_m[4][16], s_l[4][16], s_f[4][16];
  int lane = threadIdx.x & 31, wid = threadIdx.x >> 5;
  int bh = blockIdx.y;
  int b = bh >> 2, h = bh & 3;
  int q0 = (blockIdx.x * 4 + wid) * 16;
  const u16* base = qkv16 + (size_t)b * SS * 768;
  const u16* kbh = kT + (size_t)bh * HDIM * SS;
  int kh = lane >> 4;
  int rowq = q0 + (lane & 15);
  v16bf qf[2];
#pragma unroll
  for (int s2 = 0; s2 < 2; ++s2)
    qf[s2] = ldfragA(base + (size_t)rowq * 768 + h * HDIM + s2 * 32 + 8 * kh);
  v8f z = {};
  v8f acc[4];
  acc[0] = z; acc[1] = z; acc[2] = z; acc[3] = z;
  if (lane < 16) { s_m[wid][lane] = -1e30f; s_l[wid][lane] = 0.0f; }
  __syncthreads();
  for (int kt = 0; kt < SS; kt += 32) {
#pragma unroll
    for (int sub = 0; sub < 2; ++sub) {
      int n0 = kt + sub * 16;
      v8f sacc = z;
#pragma unroll
      for (int s2 = 0; s2 < 2; ++s2) {
        // B-frag: lane = key (n), elements over d (k-dim) strided in kT
        v16bf kv = ldfragB(kbh + (size_t)(s2 * 32 + 8 * kh) * SS + n0 + (lane & 15), SS);
        sacc = __builtin_amdgcn_wmma_f32_16x16x32_bf16(
            false, qf[s2], false, kv, (short)0, sacc, false, false);
      }
#pragma unroll
      for (int r = 0; r < 8; ++r)
        s_sc[wid][8 * kh + r][sub * 16 + (lane & 15)] = sacc[r] * 0.125f;
    }
    __syncthreads();
    if (lane < 16) {  // per-row online-softmax update (rows 0..15 of this wave)
      float mold = s_m[wid][lane];
      float cmax = -1e30f;
      for (int c = 0; c < 32; ++c) cmax = fmaxf(cmax, s_sc[wid][lane][c]);
      float mnew = fmaxf(mold, cmax);
      float fac = __expf(mold - mnew);
      float ssum = 0.0f;
      for (int c = 0; c < 32; ++c) {
        float e = __expf(s_sc[wid][lane][c] - mnew);
        ssum += e;
        s_p[wid][lane][c] = bf16b(e);
      }
      s_m[wid][lane] = mnew;
      s_l[wid][lane] = s_l[wid][lane] * fac + ssum;
      s_f[wid][lane] = fac;
    }
    __syncthreads();
    BF16x16 up;  // P as WMMA A-operand, read from LDS in A layout
    {
      const unsigned* pp = (const unsigned*)&s_p[wid][lane & 15][8 * kh];
#pragma unroll
      for (int i = 0; i < 4; ++i) { up.u4[i] = pp[i]; up.u4[4 + i] = pp[8 + i]; }
    }
    float fr[8];
#pragma unroll
    for (int r = 0; r < 8; ++r) fr[r] = s_f[wid][8 * kh + r];
#pragma unroll
    for (int t = 0; t < 4; ++t) {
#pragma unroll
      for (int r = 0; r < 8; ++r) acc[t][r] *= fr[r];
    }
#pragma unroll
    for (int t = 0; t < 4; ++t) {
      v16bf vv = ldfragB(base + (size_t)(kt + 8 * kh) * 768 + 512 + h * HDIM + t * 16 + (lane & 15), 768);
      acc[t] = __builtin_amdgcn_wmma_f32_16x16x32_bf16(
          false, up.v, false, vv, (short)0, acc[t], false, false);
    }
    __syncthreads();
  }
  float lr[8];
#pragma unroll
  for (int r = 0; r < 8; ++r) lr[r] = 1.0f / s_l[wid][8 * kh + r];
#pragma unroll
  for (int t = 0; t < 4; ++t) {
#pragma unroll
    for (int r = 0; r < 8; ++r) {
      int node = b * SS + q0 + 8 * kh + r;
      o[(size_t)node * CC + h * HDIM + t * 16 + (lane & 15)] = acc[t][r] * lr[r];
    }
  }
}

// ---------------------------------------------------------------------------
// Front-end: embeddings + tiny MLPs + input proj + risk gate. One block/node.
// ---------------------------------------------------------------------------
__global__ __launch_bounds__(256) void k_node_prep(
    const int* gate_id, const float* num_feat, const float* gate_emb,
    const float* np_w1, const float* np_b1, const float* np_w2, const float* np_b2,
    const float* ip_w, const float* ip_b,
    const float* rg_w1, const float* rg_b1, const float* rg_w2, const float* rg_b2,
    float* x) {
  __shared__ float s_nf[5], s_h1[32], s_cat[64], s_red[256];
  int node = blockIdx.x, tid = threadIdx.x;
  if (tid < 5) s_nf[tid] = num_feat[(size_t)node * 5 + tid];
  __syncthreads();
  if (tid < 32) {
    float a = np_b1[tid];
    for (int k = 0; k < 5; ++k) a += s_nf[k] * np_w1[k * 32 + tid];
    s_h1[tid] = fmaxf(a, 0.0f);
  }
  __syncthreads();
  if (tid < 32) {
    float a = np_b2[tid];
    for (int k = 0; k < 32; ++k) a += s_h1[k] * np_w2[k * 32 + tid];
    s_cat[32 + tid] = a;
    s_cat[tid] = gate_emb[(size_t)gate_id[node] * 32 + tid];
  }
  float rh = rg_b1[tid];
  for (int k = 0; k < 5; ++k) rh += s_nf[k] * rg_w1[k * 256 + tid];
  rh = fmaxf(rh, 0.0f);
  s_red[tid] = rh * rg_w2[tid];
  __syncthreads();
  for (int off = 128; off > 0; off >>= 1) {
    if (tid < off) s_red[tid] += s_red[tid + off];
    __syncthreads();
  }
  float risk = 1.0f / (1.0f + __expf(-(s_red[0] + rg_b2[0])));
  float a = ip_b[tid];
  for (int k = 0; k < 64; ++k) a += s_cat[k] * ip_w[k * 256 + tid];
  x[(size_t)node * CC + tid] = fmaxf(a, 0.0f) * (1.0f + risk);
}

// GIN neighborhood sum: agg[dst] += x[src]
__global__ void k_scatter_rows(const float* x, const int* src, const int* dst, float* agg) {
  long long gid = (long long)blockIdx.x * 256 + threadIdx.x;
  int e = (int)(gid >> 6);
  int c0 = ((int)gid & 63) * 4;
  int s = src[e], d = dst[e];
  const float* xp = x + (size_t)s * CC + c0;
  float* ap = agg + (size_t)d * CC + c0;
  atomAddF(ap + 0, xp[0]); atomAddF(ap + 1, xp[1]);
  atomAddF(ap + 2, xp[2]); atomAddF(ap + 3, xp[3]);
}

__global__ void k_gin_combine(const float* x, const float* agg, const float* epsp, float* h) {
  size_t i = (size_t)blockIdx.x * 256 + threadIdx.x;
  h[i] = (1.0f + epsp[0]) * x[i] + agg[i];
}

__global__ void k_init_xdir(const float* xloc, const float* fb, const float* rb, float* xdir) {
  size_t i = (size_t)blockIdx.x * 256 + threadIdx.x;
  int c = (int)(i & (CC - 1));
  xdir[i] = xloc[i] + 0.5f * (fb[c] + rb[c]);
}

// degree + self-loop edge_attr mean
__global__ void k_deg(const int* dArr, const float* eattr, float* deg, float* lsum) {
  int e = blockIdx.x * 256 + threadIdx.x;
  if (e >= EE) return;
  int d = dArr[e];
  atomAddF(deg + d, 1.0f);
  const float* ep = eattr + (size_t)e * 3;
  atomAddF(lsum + (size_t)d * 3 + 0, ep[0]);
  atomAddF(lsum + (size_t)d * 3 + 1, ep[1]);
  atomAddF(lsum + (size_t)d * 3 + 2, ep[2]);
}

__global__ void k_loopdiv(const float* deg, const float* lsum, float* latt) {
  int gid = blockIdx.x * 256 + threadIdx.x;
  if (gid >= NN * 3) return;
  latt[gid] = lsum[gid] / fmaxf(deg[gid / 3], 1.0f);
}

// GATv2 logits per (edge-or-selfloop, head) + segment max
__global__ void k_gat_logit(const int* sArr, const int* dArr, const float* eattr,
                            const float* latt, const float* xl, const float* xr,
                            const float* we, const float* att,
                            float* elog, unsigned* smax) {
  int gid = blockIdx.x * 256 + threadIdx.x;
  if (gid >= (EE + NN) * HH) return;
  int idx = gid >> 2, h = gid & 3;
  int s, d; float ea0, ea1, ea2;
  if (idx < EE) {
    s = sArr[idx]; d = dArr[idx];
    const float* ep = eattr + (size_t)idx * 3;
    ea0 = ep[0]; ea1 = ep[1]; ea2 = ep[2];
  } else {
    s = d = idx - EE;
    const float* ep = latt + (size_t)(idx - EE) * 3;
    ea0 = ep[0]; ea1 = ep[1]; ea2 = ep[2];
  }
  const float* pl = xl + (size_t)s * CC + h * HDIM;
  const float* pr = xr + (size_t)d * CC + h * HDIM;
  const float* w0 = we + h * HDIM;
  const float* w1 = we + 256 + h * HDIM;
  const float* w2 = we + 512 + h * HDIM;
  const float* pa = att + h * HDIM;
  float accv = 0.0f;
  for (int k = 0; k < HDIM; ++k) {
    float m = pl[k] + pr[k] + ea0 * w0[k] + ea1 * w1[k] + ea2 * w2[k];
    m = (m > 0.0f) ? m : LRELU_S * m;
    accv += m * pa[k];
  }
  elog[(size_t)idx * HH + h] = accv;
  atomicMax(&smax[(size_t)d * HH + h], fenc(accv));
}

__global__ void k_gat_exp(const int* dArr, float* elog, const unsigned* smax, float* den) {
  int gid = blockIdx.x * 256 + threadIdx.x;
  if (gid >= (EE + NN) * HH) return;
  int idx = gid >> 2, h = gid & 3;
  int d = (idx < EE) ? dArr[idx] : (idx - EE);
  float mx = fdec(smax[(size_t)d * HH + h]);
  float ex = __expf(elog[(size_t)idx * HH + h] - mx);
  elog[(size_t)idx * HH + h] = ex;
  atomAddF(&den[(size_t)d * HH + h], ex);
}

__global__ void k_gat_scatter(const int* sArr, const int* dArr, const float* elog,
                              const float* den, const float* xl, float* xdir) {
  int gid = blockIdx.x * 256 + threadIdx.x;
  if (gid >= (EE + NN) * 16) return;
  int idx = gid >> 4;
  int sub = gid & 15;
  int h = sub >> 2, t = sub & 3;
  int s, d;
  if (idx < EE) { s = sArr[idx]; d = dArr[idx]; } else { s = d = idx - EE; }
  float a = elog[(size_t)idx * HH + h] / (den[(size_t)d * HH + h] + 1e-16f) * 0.5f;
  const float* xp = xl + (size_t)s * CC + h * HDIM + t * 16;
  float* op = xdir + (size_t)d * CC + h * HDIM + t * 16;
#pragma unroll
  for (int i = 0; i < 16; ++i) atomAddF(op + i, a * xp[i]);
}

// in-place LayerNorm; one wave per row
__global__ __launch_bounds__(256) void k_layernorm(float* x, const float* g, const float* b) {
  int wid = threadIdx.x >> 5, lane = threadIdx.x & 31;
  int row = blockIdx.x * 8 + wid;
  float v[8]; float s = 0.0f;
#pragma unroll
  for (int i = 0; i < 8; ++i) { v[i] = x[(size_t)row * CC + lane + 32 * i]; s += v[i]; }
#pragma unroll
  for (int o = 16; o > 0; o >>= 1) s += __shfl_xor(s, o, 32);
  float mu = s * (1.0f / CC);
  float q = 0.0f;
#pragma unroll
  for (int i = 0; i < 8; ++i) { float dv = v[i] - mu; q += dv * dv; }
#pragma unroll
  for (int o = 16; o > 0; o >>= 1) q += __shfl_xor(q, o, 32);
  float rs = rsqrtf(q * (1.0f / CC) + 1e-5f);
#pragma unroll
  for (int i = 0; i < 8; ++i) {
    int c = lane + 32 * i;
    x[(size_t)row * CC + c] = (v[i] - mu) * rs * g[c] + b[c];
  }
}

// out = (a + r) * g/sqrt(1+eps) + b   (BatchNorm1d eval, running stats 0/1)
__global__ void k_bn_add(const float* a, const float* r, const float* g, const float* b,
                         float* out) {
  size_t i = (size_t)blockIdx.x * 256 + threadIdx.x;
  int c = (int)(i & (CC - 1));
  float sc = g[c] * rsqrtf(1.0f + 1e-5f);
  out[i] = (a[i] + r[i]) * sc + b[c];
}

__global__ void k_head_bin(const float* hid, const float* w2, const float* b2, float* out) {
  int n = blockIdx.x * 256 + threadIdx.x;
  const float* hp = hid + (size_t)n * 128;
  float a = b2[0];
  for (int k = 0; k < 128; ++k) a += hp[k] * w2[k];
  out[n] = a;
}

__global__ void k_head_type(const float* hid, const float* w2, const float* b2, float* out) {
  int gid = blockIdx.x * 256 + threadIdx.x;
  if (gid >= NN * 5) return;
  int n = gid / 5, t = gid % 5;
  const float* hp = hid + (size_t)n * 128;
  float a = b2[t];
  for (int k = 0; k < 128; ++k) a += hp[k] * w2[k * 5 + t];
  out[(size_t)n * 5 + t] = a;
}

// ---------------------------------------------------------------------------
static inline void cvt(const float* in, u16* out, long long n, hipStream_t st) {
  k_cvt<<<(unsigned)((n / 8 + 255) / 256), 256, 0, st>>>(in, out, n);
}
static inline void gemm(const u16* A16, int lda, const u16* W16, int ldw,
                        const float* bias, float* C, int ldc,
                        int M, int K, int Nc, int flags, hipStream_t st) {
  dim3 g((unsigned)(M / 64), (unsigned)(Nc / 64));
  k_gemm<<<g, 128, 0, st>>>(A16, lda, W16, ldw, bias, C, ldc, M, K, Nc, flags);
}

static void run_gat(hipStream_t st, const u16* xloc16, float* xl, float* xr, float* xdir,
                    const int* sArr, const int* dArr, const float* eattr,
                    const u16* wl16, const float* bl, const u16* wr16, const float* br,
                    const float* we, const float* att,
                    float* deg, float* lsum, float* latt, unsigned* smax, float* den,
                    float* elog) {
  hipMemsetAsync(deg, 0, NN * sizeof(float), st);
  hipMemsetAsync(lsum, 0, (size_t)NN * 3 * sizeof(float), st);
  hipMemsetAsync(smax, 0, (size_t)NN * HH * sizeof(unsigned), st);
  hipMemsetAsync(den, 0, (size_t)NN * HH * sizeof(float), st);
  k_deg<<<EE / 256, 256, 0, st>>>(dArr, eattr, deg, lsum);
  k_loopdiv<<<(NN * 3 + 255) / 256, 256, 0, st>>>(deg, lsum, latt);
  gemm(xloc16, CC, wl16, CC, bl, xl, CC, NN, CC, CC, 0, st);
  gemm(xloc16, CC, wr16, CC, br, xr, CC, NN, CC, CC, 0, st);
  int totL = (EE + NN) * HH;
  k_gat_logit<<<(totL + 255) / 256, 256, 0, st>>>(sArr, dArr, eattr, latt, xl, xr, we, att,
                                                  elog, smax);
  k_gat_exp<<<(totL + 255) / 256, 256, 0, st>>>(dArr, elog, smax, den);
  int totS = (EE + NN) * 16;
  k_gat_scatter<<<(totS + 255) / 256, 256, 0, st>>>(sArr, dArr, elog, den, xl, xdir);
}

extern "C" void kernel_launch(void* const* d_in, const int* in_sizes, int n_in,
                              void* d_out, int out_size, void* d_ws, size_t ws_size,
                              hipStream_t stream) {
  (void)in_sizes; (void)n_in; (void)out_size; (void)ws_size;
  const int*   gate_id  = (const int*)d_in[0];
  const float* num_feat = (const float*)d_in[1];
  const int*   eidx     = (const int*)d_in[2];
  const float* eattr    = (const float*)d_in[3];
  // d_in[4] = batch (implied by sorted equal-size layout)
  const float* P[54];
  for (int i = 0; i < 54; ++i) P[i] = (const float*)d_in[5 + i];
  const float *gate_emb = P[0];
  const float *np_w1 = P[1],  *np_b1 = P[2],  *np_w2 = P[3],  *np_b2 = P[4];
  const float *ip_w = P[5],   *ip_b = P[6];
  const float *rg_w1 = P[7],  *rg_b1 = P[8],  *rg_w2 = P[9],  *rg_b2 = P[10];
  const float *gin_eps = P[11];
  const float *gin_w1 = P[12], *gin_b1 = P[13], *gin_w2 = P[14], *gin_b2 = P[15];
  const float *f_wl = P[16], *f_bl = P[17], *f_wr = P[18], *f_br = P[19];
  const float *f_we = P[20], *f_att = P[21], *f_bias = P[22];
  const float *r_wl = P[23], *r_bl = P[24], *r_wr = P[25], *r_br = P[26];
  const float *r_we = P[27], *r_att = P[28], *r_bias = P[29];
  const float *ln_g = P[30], *ln_b = P[31];
  const float *mha_wqkv = P[32], *mha_bqkv = P[33], *mha_wo = P[34], *mha_bo = P[35];
  const float *bn2_g = P[36], *bn2_b = P[37];
  const float *mlp_w1 = P[38], *mlp_b1 = P[39], *mlp_w2 = P[40], *mlp_b2 = P[41];
  const float *bn3_g = P[42], *bn3_b = P[43];
  const float *fuse_w = P[44], *fuse_b = P[45];
  const float *bh_w1 = P[46], *bh_b1 = P[47], *bh_w2 = P[48], *bh_b2 = P[49];
  const float *th_w1 = P[50], *th_b1 = P[51], *th_w2 = P[52], *th_b2 = P[53];

  float* w = (float*)d_ws;
  const size_t F = (size_t)NN * CC;  // 4,194,304 floats
  float* xbuf = w;            // x; later qkv[0:F); later hg
  float* hbuf = w + F;        // agg/h; later qkv/mlp-hidden
  float* ginh = w + 2 * F;    // GIN hidden; later qkv/mlp-hidden
  float* xloc = w + 3 * F;    // live to end
  float* xdir = w + 4 * F;    // live to end
  float* xglb = w + 5 * F;    // live to end
  float* xlb  = w + 6 * F;    // xl / attnO / mlp-out / head-hidden
  float* xrb  = w + 7 * F;    // xr / o-proj
  float* elog = w + 8 * F;                               // (E+N)*H
  float* smaxf = elog + (size_t)(EE + NN) * HH;          // N*H (as unsigned)
  float* den  = smaxf + (size_t)NN * HH;                 // N*H
  float* deg  = den + (size_t)NN * HH;                   // N
  float* lsum = deg + NN;                                // 3N
  float* latt = lsum + (size_t)NN * 3;                   // 3N
  // bf16 weight arena (persistent within one launch)
  u16* wb = (u16*)(latt + (size_t)NN * 3);
  u16* w16_gin1 = wb;                 // 65536
  u16* w16_gin2 = wb + 65536;         // 65536
  u16* w16_fwl  = wb + 131072;        // 65536
  u16* w16_fwr  = wb + 196608;        // 65536
  u16* w16_rwl  = wb + 262144;        // 65536
  u16* w16_rwr  = wb + 327680;        // 65536
  u16* w16_qkv  = wb + 393216;        // 196608
  u16* w16_wo   = wb + 589824;        // 65536
  u16* w16_mlp1 = wb + 655360;        // 131072
  u16* w16_mlp2 = wb + 786432;        // 131072
  u16* w16_fuse = wb + 917504;        // 196608
  u16* w16_bh1  = wb + 1114112;       // 32768
  u16* w16_th1  = wb + 1146880;       // 32768  (arena total 1,179,648 u16)
  // bf16 activation staging (reused between GEMMs; max = qkv16 = N*768)
  u16* ab16 = wb + 1179648;
  // transposed K for flash attention: [(b,h,d), key] = 4M u16
  u16* kT = ab16 + (size_t)NN * 768;
  float* qkv  = w;            // [0, 3F)
  float* hg   = w;            // after flash consumes qkv
  float* mlph = w + F;        // [F, 3F)
  float* attnO = xlb;
  float* oproj = xrb;
  float* mlpo  = xlb;
  float* hid   = xlb;
  float* out = (float*)d_out;
  float* bin = out;
  float* typl = out + NN;
  float* emb = out + (size_t)NN * 6;
  const int* srcA = eidx;
  const int* dstA = eidx + EE;

  // --- weights -> bf16 (once per launch; tiny, L2 resident) ---
  cvt(gin_w1, w16_gin1, 65536, stream);
  cvt(gin_w2, w16_gin2, 65536, stream);
  cvt(f_wl, w16_fwl, 65536, stream);
  cvt(f_wr, w16_fwr, 65536, stream);
  cvt(r_wl, w16_rwl, 65536, stream);
  cvt(r_wr, w16_rwr, 65536, stream);
  cvt(mha_wqkv, w16_qkv, 196608, stream);
  cvt(mha_wo, w16_wo, 65536, stream);
  cvt(mlp_w1, w16_mlp1, 131072, stream);
  cvt(mlp_w2, w16_mlp2, 131072, stream);
  cvt(fuse_w, w16_fuse, 196608, stream);
  cvt(bh_w1, w16_bh1, 32768, stream);
  cvt(th_w1, w16_th1, 32768, stream);

  // --- front-end ---
  hipMemsetAsync(hbuf, 0, F * sizeof(float), stream);
  k_node_prep<<<NN, 256, 0, stream>>>(gate_id, num_feat, gate_emb,
                                      np_w1, np_b1, np_w2, np_b2, ip_w, ip_b,
                                      rg_w1, rg_b1, rg_w2, rg_b2, xbuf);
  k_scatter_rows<<<(unsigned)((size_t)EE * 64 / 256), 256, 0, stream>>>(xbuf, srcA, dstA, hbuf);
  k_gin_combine<<<NN * CC / 256, 256, 0, stream>>>(xbuf, hbuf, gin_eps, hbuf);
  cvt(hbuf, ab16, (long long)F, stream);
  gemm(ab16, CC, w16_gin1, CC, gin_b1, ginh, CC, NN, CC, CC, GF_RELU, stream);
  cvt(ginh, ab16, (long long)F, stream);
  gemm(ab16, CC, w16_gin2, CC, gin_b2, xloc, CC, NN, CC, CC, 0, stream);

  // --- bidirectional GATv2 + residual + LN ---
  k_init_xdir<<<NN * CC / 256, 256, 0, stream>>>(xloc, f_bias, r_bias, xdir);
  cvt(xloc, ab16, (long long)F, stream);   // xloc16 reused by both directions
  run_gat(stream, ab16, xlb, xrb, xdir, srcA, dstA, eattr,
          w16_fwl, f_bl, w16_fwr, f_br, f_we, f_att,
          deg, lsum, latt, (unsigned*)smaxf, den, elog);
  run_gat(stream, ab16, xlb, xrb, xdir, dstA, srcA, eattr,
          w16_rwl, r_bl, w16_rwr, r_br, r_we, r_att,
          deg, lsum, latt, (unsigned*)smaxf, den, elog);
  k_layernorm<<<NN / 8, 256, 0, stream>>>(xdir, ln_g, ln_b);

  // --- GPS: dense per-graph MHA + BN + FFN ---
  cvt(xdir, ab16, (long long)F, stream);
  gemm(ab16, CC, w16_qkv, 768, mha_bqkv, qkv, 768, NN, CC, 768, 0, stream);
  cvt(qkv, ab16, (long long)NN * 768, stream);
  k_ktrans<<<(BB * HH * HDIM * (SS / 8)) / 256, 256, 0, stream>>>(ab16, kT);
  dim3 fg(SS / 64, BB * HH);
  k_flash<<<fg, 128, 0, stream>>>(ab16, kT, attnO);
  cvt(attnO, ab16, (long long)F, stream);
  gemm(ab16, CC, w16_wo, CC, mha_bo, oproj, CC, NN, CC, CC, 0, stream);
  k_bn_add<<<NN * CC / 256, 256, 0, stream>>>(oproj, xdir, bn2_g, bn2_b, hg);
  cvt(hg, ab16, (long long)F, stream);
  gemm(ab16, CC, w16_mlp1, 512, mlp_b1, mlph, 512, NN, CC, 512, GF_RELU, stream);
  cvt(mlph, ab16, (long long)NN * 512, stream);
  gemm(ab16, 512, w16_mlp2, CC, mlp_b2, mlpo, CC, NN, 512, CC, 0, stream);
  k_bn_add<<<NN * CC / 256, 256, 0, stream>>>(hg, mlpo, bn3_g, bn3_b, xglb);

  // --- fuse (concat GEMM via accumulate chain) + heads ---
  cvt(xloc, ab16, (long long)F, stream);
  gemm(ab16, CC, w16_fuse, CC, fuse_b, emb, CC, NN, CC, CC, 0, stream);
  cvt(xdir, ab16, (long long)F, stream);
  gemm(ab16, CC, w16_fuse + 256 * 256, CC, nullptr, emb, CC, NN, CC, CC, GF_ACCUM, stream);
  cvt(xglb, ab16, (long long)F, stream);
  gemm(ab16, CC, w16_fuse + 512 * 256, CC, nullptr, emb, CC, NN, CC, CC,
       GF_ACCUM | GF_RELU, stream);
  cvt(emb, ab16, (long long)F, stream);
  gemm(ab16, CC, w16_bh1, 128, bh_b1, hid, 128, NN, CC, 128, GF_RELU, stream);
  k_head_bin<<<NN / 256, 256, 0, stream>>>(hid, bh_w2, bh_b2, bin);
  gemm(ab16, CC, w16_th1, 128, th_b1, hid, 128, NN, CC, 128, GF_RELU, stream);
  k_head_type<<<(NN * 5 + 255) / 256, 256, 0, stream>>>(hid, th_w2, th_b2, typl);
}
